// Matcher_84576495992874
// MI455X (gfx1250) — matrix operand order repaired and checked by reading, compiled, not run on hardware
//
#include <hip/hip_runtime.h>
#include <hip/hip_bf16.h>
#include <stdint.h>

// Problem constants (from reference): OBJ_N=3, D_KEY=128, D_VAL=512, BANK_N=8192, Q_N=4096, B=1
#define OBJ_N  3
#define DKEY   128
#define DVAL   512
#define BANKN  8192
#define QN     4096
#define QT     32          // queries per workgroup (2 WMMA N-tiles)

typedef __attribute__((ext_vector_type(16))) __bf16          v16bf;
typedef __attribute__((ext_vector_type(8)))  float           v8f;
typedef __attribute__((ext_vector_type(8)))  unsigned short  ushort8;

union BF16x16 { v16bf bf; ushort8 u[2]; };

// f32 -> bf16 (round-to-nearest-even), pure integer ops
__device__ __forceinline__ unsigned short f2bf(float f) {
    unsigned int u = __float_as_uint(f);
    u += 0x7FFFu + ((u >> 16) & 1u);
    return (unsigned short)(u >> 16);
}

// ---------------- Pre-pass kernels: convert / transpose into bf16 workspace ----------------

// keysT[o][bank][k] = keys[o][k][bank]
__global__ void conv_keysT(const float* __restrict__ keys, unsigned short* __restrict__ keysT) {
    int idx = blockIdx.x * blockDim.x + threadIdx.x;          // [0, 3*128*8192)
    int o    = idx / (DKEY * BANKN);
    int rem  = idx % (DKEY * BANKN);
    int k    = rem / BANKN;
    int bank = rem % BANKN;
    keysT[(size_t)o * BANKN * DKEY + (size_t)bank * DKEY + k] = f2bf(keys[idx]);
}

// valsB[i] = bf16(values[i])   (layout [o][v][bank] already has K=bank contiguous)
__global__ void conv_vals(const float* __restrict__ vals, unsigned short* __restrict__ valsB) {
    int idx = blockIdx.x * blockDim.x + threadIdx.x;          // [0, 3*512*8192)
    valsB[idx] = f2bf(vals[idx]);
}

// qinT[q][k] = q_in[k][q]
__global__ void conv_qinT(const float* __restrict__ qin, unsigned short* __restrict__ qinT) {
    int idx = blockIdx.x * blockDim.x + threadIdx.x;          // [0, 128*4096)
    int k = idx / QN;
    int q = idx % QN;
    qinT[(size_t)q * DKEY + k] = f2bf(qin[idx]);
}

// out[0][o][512+v][q] = q_out[v][q]  (pass-through rows, broadcast over objects)
__global__ void copy_qout(const float* __restrict__ qout, float* __restrict__ out) {
    int idx = blockIdx.x * blockDim.x + threadIdx.x;          // [0, 3*512*4096)
    int o   = idx / (DVAL * QN);
    int rem = idx % (DVAL * QN);
    out[(size_t)o * (DVAL + DVAL) * QN + (size_t)DVAL * QN + rem] = qout[rem];
}

// ---------------- Fused flash-attention kernel ----------------
// grid = (QN/32, OBJ_N), block = 256 (8 wave32)
// Each workgroup: one object, one 32-query tile (2 WMMA N-tiles), loops over 8192 banks
// in blocks of 128. Wave w owns score-tile banks [w*16, w*16+16) of each block, and
// V rows [w*64, w*64+64). Each A fragment (keys / values) is reused across both N-tiles.

__global__ __launch_bounds__(256)
void flash_attn(const unsigned short* __restrict__ keysT,
                const unsigned short* __restrict__ valsB,
                const unsigned short* __restrict__ qinT,
                float* __restrict__ out) {
    const int o     = blockIdx.y;
    const int qt    = blockIdx.x;
    const int tid   = threadIdx.x;
    const int w     = tid >> 5;        // wave id 0..7
    const int lane  = tid & 31;
    const int lhalf = lane >> 4;       // 0 / 1
    const int l16   = lane & 15;       // N index within a 16-wide tile

    // 1/sqrt(128) * log2(e): work in exp2 domain -> v_exp_f32 directly
    const float SCALE = 0.08838834764831845f * 1.4426950408889634f;

    __shared__ __attribute__((aligned(16))) unsigned short Plds[QT * 128]; // P[q][bank] bf16
    __shared__ float wred[8][QT];                                          // per-wave reductions

    // ---- preload stage-1 B fragments: q_in, 4 K-steps x 2 N-tiles ----
    BF16x16 Bq[4][2];
#pragma unroll
    for (int n = 0; n < 2; ++n) {
        const unsigned short* qrow = qinT + (size_t)(qt * QT + n * 16 + l16) * DKEY;
#pragma unroll
        for (int ks = 0; ks < 4; ++ks) {
            int kb = ks * 32 + lhalf * 16;        // B: lanes 0-15 K 0..15; lanes 16-31 K 16..31
            Bq[ks][n].u[0] = *(const ushort8*)(qrow + kb);
            Bq[ks][n].u[1] = *(const ushort8*)(qrow + kb + 8);
        }
    }

    v8f acc[4][2];
#pragma unroll
    for (int t = 0; t < 4; ++t) { acc[t][0] = (v8f)0.0f; acc[t][1] = (v8f)0.0f; }
    float m0 = -__builtin_inff(), m1 = -__builtin_inff();
    float l0 = 0.0f, l1 = 0.0f;

    const unsigned short* kbase = keysT + (size_t)o * BANKN * DKEY;
    const unsigned short* vbase = valsB + (size_t)o * DVAL * BANKN;

    for (int b0 = 0; b0 < BANKN; b0 += 128) {
        // prefetch next bank-block of keys for this wave's sub-tile
        {
            int nb = (b0 + 128 < BANKN) ? (b0 + 128) : b0;
            __builtin_prefetch(kbase + (size_t)(nb + w * 16 + l16) * DKEY, 0, 0);
        }

        // ---- stage 1: S[16 banks x 32 q] = keys_tile * q_tile (A reused for both N-tiles) ----
        const unsigned short* krow = kbase + (size_t)(b0 + w * 16 + l16) * DKEY;
        v8f S0 = (v8f)0.0f, S1 = (v8f)0.0f;
#pragma unroll
        for (int ks = 0; ks < 4; ++ks) {
            BF16x16 A;
            int kb = ks * 32 + lhalf * 16;        // A: lanes 0-15 K 0..15; lanes 16-31 K 16..31
            A.u[0] = *(const ushort8*)(krow + kb);
            A.u[1] = *(const ushort8*)(krow + kb + 8);
            S0 = __builtin_amdgcn_wmma_f32_16x16x32_bf16(false, A.bf, false, Bq[ks][0].bf,
                                                         (short)0, S0, false, false);
            S1 = __builtin_amdgcn_wmma_f32_16x16x32_bf16(false, A.bf, false, Bq[ks][1].bf,
                                                         (short)0, S1, false, false);
        }
#pragma unroll
        for (int e = 0; e < 8; ++e) { S0[e] *= SCALE; S1[e] *= SCALE; }

        // ---- online softmax: block max over the 128 banks, per query ----
        float tm0 = S0[0], tm1 = S1[0];
#pragma unroll
        for (int e = 1; e < 8; ++e) { tm0 = fmaxf(tm0, S0[e]); tm1 = fmaxf(tm1, S1[e]); }
        tm0 = fmaxf(tm0, __shfl_xor(tm0, 16, 32));   // combine M 0-7 with 8-15
        tm1 = fmaxf(tm1, __shfl_xor(tm1, 16, 32));
        if (lane < 16) { wred[w][l16] = tm0; wred[w][16 + l16] = tm1; }
        __syncthreads();                             // also fences prior stage-2 Plds reads
        float bm0 = wred[0][l16], bm1 = wred[0][16 + l16];
#pragma unroll
        for (int ww = 1; ww < 8; ++ww) {
            bm0 = fmaxf(bm0, wred[ww][l16]);
            bm1 = fmaxf(bm1, wred[ww][16 + l16]);
        }

        float mn0 = fmaxf(m0, bm0), mn1 = fmaxf(m1, bm1);
        float rs0 = exp2f(m0 - mn0), rs1 = exp2f(m1 - mn1);  // 0 on first block
        m0 = mn0; m1 = mn1;

        // ---- exponentiate, pack bf16 P to LDS, partial row-sums ----
        float ps0 = 0.0f, ps1 = 0.0f;
        ushort8 pk0, pk1;
#pragma unroll
        for (int e = 0; e < 8; ++e) {
            float p0 = exp2f(S0[e] - mn0);
            float p1 = exp2f(S1[e] - mn1);
            ps0 += p0; ps1 += p1;
            pk0[e] = f2bf(p0); pk1[e] = f2bf(p1);
        }
        // lane's 8 banks are contiguous: block-bank = w*16 + lhalf*8 + e
        *(ushort8*)(Plds + (l16)      * 128 + w * 16 + lhalf * 8) = pk0;
        *(ushort8*)(Plds + (16 + l16) * 128 + w * 16 + lhalf * 8) = pk1;
        ps0 += __shfl_xor(ps0, 16, 32);
        ps1 += __shfl_xor(ps1, 16, 32);
        __syncthreads();                             // wred max-phase done before sum-phase
        if (lane < 16) { wred[w][l16] = ps0; wred[w][16 + l16] = ps1; }
        __syncthreads();
        float bs0 = 0.0f, bs1 = 0.0f;
#pragma unroll
        for (int ww = 0; ww < 8; ++ww) { bs0 += wred[ww][l16]; bs1 += wred[ww][16 + l16]; }

        l0 = l0 * rs0 + bs0;
        l1 = l1 * rs1 + bs1;
#pragma unroll
        for (int t = 0; t < 4; ++t)
#pragma unroll
            for (int e = 0; e < 8; ++e) { acc[t][0][e] *= rs0; acc[t][1][e] *= rs1; }

        // ---- stage 2: acc += V_tile * P  (A fragment reused across both N-tiles) ----
#pragma unroll
        for (int ks = 0; ks < 4; ++ks) {
            BF16x16 Bp0, Bp1;
            int kb = ks * 32 + lhalf * 16;
            Bp0.u[0] = *(const ushort8*)(Plds + (l16)      * 128 + kb);
            Bp0.u[1] = *(const ushort8*)(Plds + (l16)      * 128 + kb + 8);
            Bp1.u[0] = *(const ushort8*)(Plds + (16 + l16) * 128 + kb);
            Bp1.u[1] = *(const ushort8*)(Plds + (16 + l16) * 128 + kb + 8);
#pragma unroll
            for (int t = 0; t < 4; ++t) {
                const unsigned short* vrow =
                    vbase + (size_t)(w * 64 + t * 16 + l16) * BANKN;
                BF16x16 Av;
                int kbb = b0 + ks * 32 + lhalf * 16;
                Av.u[0] = *(const ushort8*)(vrow + kbb);
                Av.u[1] = *(const ushort8*)(vrow + kbb + 8);
                acc[t][0] = __builtin_amdgcn_wmma_f32_16x16x32_bf16(false, Av.bf, false, Bp0.bf,
                                                                    (short)0, acc[t][0], false, false);
                acc[t][1] = __builtin_amdgcn_wmma_f32_16x16x32_bf16(false, Av.bf, false, Bp1.bf,
                                                                    (short)0, acc[t][1], false, false);
            }
        }
        __syncthreads();                             // protect Plds/wred before next block
    }

    // ---- finalize: divide by softmax denominator, write mem rows ----
    float inv0 = 1.0f / l0, inv1 = 1.0f / l1;
#pragma unroll
    for (int t = 0; t < 4; ++t)
#pragma unroll
        for (int e = 0; e < 8; ++e) {
            int row = w * 64 + t * 16 + lhalf * 8 + e;   // C layout: VGPR e -> M e / 8+e
            size_t base = ((size_t)o * (DVAL + DVAL) + row) * QN + qt * QT + l16;
            out[base]      = acc[t][0][e] * inv0;
            out[base + 16] = acc[t][1][e] * inv1;
        }
}

// ---------------- Launcher ----------------

extern "C" void kernel_launch(void* const* d_in, const int* in_sizes, int n_in,
                              void* d_out, int out_size, void* d_ws, size_t ws_size,
                              hipStream_t stream) {
    (void)in_sizes; (void)n_in; (void)out_size; (void)ws_size;

    const float* keys = (const float*)d_in[0];   // [3,128,8192]
    const float* vals = (const float*)d_in[1];   // [3,512,8192]
    const float* qin  = (const float*)d_in[2];   // [1,128,4096]
    const float* qout = (const float*)d_in[3];   // [1,512,4096]
    float* out = (float*)d_out;                  // [1,3,1024,4096]

    unsigned short* keysT = (unsigned short*)d_ws;                       // 3*8192*128
    unsigned short* valsB = keysT + (size_t)OBJ_N * BANKN * DKEY;        // 3*512*8192
    unsigned short* qinT  = valsB + (size_t)OBJ_N * DVAL * BANKN;        // 4096*128

    conv_keysT<<<(OBJ_N * DKEY * BANKN) / 256, 256, 0, stream>>>(keys, keysT);
    conv_vals <<<(OBJ_N * DVAL * BANKN) / 256, 256, 0, stream>>>(vals, valsB);
    conv_qinT <<<(DKEY * QN)           / 256, 256, 0, stream>>>(qin,  qinT);
    copy_qout <<<(OBJ_N * DVAL * QN)   / 256, 256, 0, stream>>>(qout, out);

    dim3 grid(QN / QT, OBJ_N);
    flash_attn<<<grid, 256, 0, stream>>>(keysT, valsB, qinT, out);
}
